// KMeansLayer_35931696399011
// MI455X (gfx1250) — matrix operand order repaired and checked by reading, not compiled, not run on hardware
//
#include <hip/hip_runtime.h>
#include <hip/hip_bf16.h>

typedef _Float16 v16h __attribute__((ext_vector_type(16)));
typedef _Float16 v8h  __attribute__((ext_vector_type(8)));
typedef float    v8f  __attribute__((ext_vector_type(8)));

#define DIM      256
#define NCLUST   1024
#define NITER    10
#define NROUND   (NCLUST / 32)   // 32 rounds of 32 clusters
#define ROWS_PER_BLOCK 128       // 8 waves * 16 rows
#define BPAD     8               // LDS row pad (halfs) -> 264-half row stride
#define BROW     (DIM + BPAD)    // 264

// ---------------------------------------------------------------------------
// One-time: data f32 -> f16, and x2[i] = sum_k data[i][k]^2
// ---------------------------------------------------------------------------
__global__ __launch_bounds__(256) void preprocess_kernel(
    const float* __restrict__ data, _Float16* __restrict__ dh, float* __restrict__ x2)
{
    const int i = blockIdx.x;
    const int k = threadIdx.x;
    float v = data[(size_t)i * DIM + k];
    dh[(size_t)i * DIM + k] = (_Float16)v;
    __shared__ float red[256];
    red[k] = v * v;
    __syncthreads();
    for (int s = 128; s > 0; s >>= 1) {
        if (k < s) red[k] += red[k + s];
        __syncthreads();
    }
    if (k == 0) x2[i] = red[0];
}

// ---------------------------------------------------------------------------
// Deterministic init: c[j] = data[j * 97]
// ---------------------------------------------------------------------------
__global__ __launch_bounds__(256) void init_centers_kernel(
    const float* __restrict__ data, float* __restrict__ c)
{
    const int j = blockIdx.x;
    const int k = threadIdx.x;
    c[(size_t)j * DIM + k] = data[(size_t)(j * 97) * DIM + k];
}

// ---------------------------------------------------------------------------
// Per-iteration: centers f32 -> f16, and c2[j] = ||c_j||^2
// ---------------------------------------------------------------------------
__global__ __launch_bounds__(256) void prep_centers_kernel(
    const float* __restrict__ c, _Float16* __restrict__ ch, float* __restrict__ c2)
{
    const int j = blockIdx.x;
    const int k = threadIdx.x;
    float v = c[(size_t)j * DIM + k];
    ch[(size_t)j * DIM + k] = (_Float16)v;
    __shared__ float red[256];
    red[k] = v * v;
    __syncthreads();
    for (int s = 128; s > 0; s >>= 1) {
        if (k < s) red[k] += red[k + s];
        __syncthreads();
    }
    if (k == 0) c2[j] = red[0];
}

// ---------------------------------------------------------------------------
// Load one 16x16x32 B fragment from the padded LDS tile.
// ---------------------------------------------------------------------------
__device__ __forceinline__ v16h lds_bfrag(const _Float16* bt, int kk) {
    v16h b;
    *((v8h*)&b)       = *(const v8h*)(bt + kk * 32);
    *(((v8h*)&b) + 1) = *(const v8h*)(bt + kk * 32 + 16);
    return b;
}

// ---------------------------------------------------------------------------
// Assignment: a[i] = argmin_j (c2[j] - 2 * x_i . c_j)
// B staged in LDS (double buffered, shared by 8 waves); A cached in registers;
// two independent WMMA chains; B fragments software-pipelined one K-step ahead.
// ---------------------------------------------------------------------------
__global__ __launch_bounds__(256) void assign_kernel(
    const _Float16* __restrict__ Ah,   // data f16, N x 256
    const _Float16* __restrict__ Bh,   // centers f16, 1024 x 256
    const float*    __restrict__ c2,   // 1024
    int*            __restrict__ a,    // N
    int N)
{
    __shared__ _Float16 Bs[2][32 * BROW];   // 2 x 16.5KB
    __shared__ float    c2s[NCLUST];        // 4KB

    const int lane = threadIdx.x & 31;
    const int wave = threadIdx.x >> 5;
    const int rb0  = blockIdx.x * ROWS_PER_BLOCK + wave * 16;
    const bool active = rb0 < N;
    const int rowbase = active ? rb0 : 0;   // redundant waves compute row 0, don't write
    const int h   = lane >> 4;
    const int l16 = lane & 15;

    // staging role (all 256 threads): row = t>>3, 32-half segment = t&7
    const int srow = threadIdx.x >> 3;
    const int sseg = (threadIdx.x & 7) * 32;

    // A fragments per ISA layout
    v16h afrag[8];
    {
        const _Float16* arow = Ah + (size_t)(rowbase + l16) * DIM;
        #pragma unroll
        for (int kk = 0; kk < 8; kk++) {
            const int k0 = kk * 32 + h * 8;
            v16h f;
            *((v8h*)&f)       = *(const v8h*)(arow + k0);
            *(((v8h*)&f) + 1) = *(const v8h*)(arow + k0 + 16);
            afrag[kk] = f;
        }
    }

    // stage round 0 + c2 into LDS
    {
        const _Float16* src = Bh + (size_t)srow * DIM + sseg;
        _Float16* dst = &Bs[0][srow * BROW + sseg];
        #pragma unroll
        for (int cch = 0; cch < 4; cch++)
            *(v8h*)(dst + cch * 8) = *(const v8h*)(src + cch * 8);
        #pragma unroll
        for (int t = 0; t < 4; t++)
            c2s[threadIdx.x + t * 256] = c2[threadIdx.x + t * 256];
    }
    __syncthreads();

    float bestv0[8], bestv1[8];
    int   besti0[8], besti1[8];
    #pragma unroll
    for (int v = 0; v < 8; v++) {
        bestv0[v] = 3.4e38f; besti0[v] = 0;
        bestv1[v] = 3.4e38f; besti1[v] = 0;
    }

    for (int r = 0; r < NROUND; r++) {
        const int cur = r & 1;
        const int base = r * 32;

        // issue next tile's global loads BEFORE compute (latency hidden by WMMAs)
        v8h g[4];
        if (r + 1 < NROUND) {
            const _Float16* src = Bh + (size_t)((r + 1) * 32 + srow) * DIM + sseg;
            #pragma unroll
            for (int cch = 0; cch < 4; cch++)
                g[cch] = *(const v8h*)(src + cch * 8);
        }

        // two interleaved WMMA chains, fragments pipelined one K-step ahead
        const _Float16* bt0 = &Bs[cur][(unsigned)l16 * BROW + h * 8];
        const _Float16* bt1 = bt0 + 16 * BROW;
        v8f acc0 = {}, acc1 = {};
        v16h b0 = lds_bfrag(bt0, 0);
        v16h b1 = lds_bfrag(bt1, 0);
        #pragma unroll
        for (int kk = 0; kk < 8; kk++) {
            v16h n0, n1;
            if (kk < 7) {
                n0 = lds_bfrag(bt0, kk + 1);
                n1 = lds_bfrag(bt1, kk + 1);
            }
            acc0 = __builtin_amdgcn_wmma_f32_16x16x32_f16(
                false, afrag[kk], false, b0, (short)0, acc0, false, false);
            acc1 = __builtin_amdgcn_wmma_f32_16x16x32_f16(
                false, afrag[kk], false, b1, (short)0, acc1, false, false);
            b0 = n0; b1 = n1;
        }

        // epilogue: lane holds (row = v + 8*h, col = l16) of each 16x16 subtile
        const int   col0 = base + l16;
        const int   col1 = base + 16 + l16;
        const float c2v0 = c2s[col0];
        const float c2v1 = c2s[col1];
        #pragma unroll
        for (int v = 0; v < 8; v++) {
            float s0 = __builtin_fmaf(-2.0f, acc0[v], c2v0);
            float s1 = __builtin_fmaf(-2.0f, acc1[v], c2v1);
            if (s0 < bestv0[v]) { bestv0[v] = s0; besti0[v] = col0; }
            if (s1 < bestv1[v]) { bestv1[v] = s1; besti1[v] = col1; }
        }

        // park next tile into LDS, then barrier
        if (r + 1 < NROUND) {
            _Float16* dst = &Bs[(r + 1) & 1][srow * BROW + sseg];
            #pragma unroll
            for (int cch = 0; cch < 4; cch++)
                *(v8h*)(dst + cch * 8) = g[cch];
        }
        __syncthreads();
    }

    // merge subtile trackers, then reduce over the 16 lanes of each half
    #pragma unroll
    for (int v = 0; v < 8; v++) {
        if (bestv1[v] < bestv0[v] ||
            (bestv1[v] == bestv0[v] && besti1[v] < besti0[v])) {
            bestv0[v] = bestv1[v]; besti0[v] = besti1[v];
        }
    }
    #pragma unroll
    for (int m = 1; m < 16; m <<= 1) {
        #pragma unroll
        for (int v = 0; v < 8; v++) {
            float ov = __shfl_xor(bestv0[v], m, 32);
            int   oi = __shfl_xor(besti0[v], m, 32);
            if (ov < bestv0[v] || (ov == bestv0[v] && oi < besti0[v])) {
                bestv0[v] = ov; besti0[v] = oi;
            }
        }
    }
    if (active && l16 == 0) {
        #pragma unroll
        for (int v = 0; v < 8; v++) a[rowbase + h * 8 + v] = besti0[v];
    }
}

// ---------------------------------------------------------------------------
// Segment sums: sums[a[i]] += data[i]; counts[a[i]] += 1
// ---------------------------------------------------------------------------
__global__ __launch_bounds__(256) void accumulate_kernel(
    const float* __restrict__ data, const int* __restrict__ a,
    float* __restrict__ sums, float* __restrict__ counts)
{
    const int i = blockIdx.x;
    const int k = threadIdx.x;
    const int cl = a[i];
    if (k == 0) atomicAdd(&counts[cl], 1.0f);
    atomicAdd(&sums[(size_t)cl * DIM + k], data[(size_t)i * DIM + k]);
}

// ---------------------------------------------------------------------------
// Center update: mean where count>0, keep old center for dead clusters
// ---------------------------------------------------------------------------
__global__ __launch_bounds__(256) void update_kernel(
    const float* __restrict__ sums, const float* __restrict__ counts,
    float* __restrict__ c)
{
    const int j = blockIdx.x;
    const int k = threadIdx.x;
    const float cnt = counts[j];
    if (cnt > 0.0f) c[(size_t)j * DIM + k] = sums[(size_t)j * DIM + k] / cnt;
}

// ---------------------------------------------------------------------------
// Final: index[j] = argmin_i (x2[i] - 2 x_i . c_j), packed u64 atomic-min
// ---------------------------------------------------------------------------
__global__ __launch_bounds__(256) void index_kernel(
    const _Float16* __restrict__ Ah,
    const _Float16* __restrict__ Bh,
    const float*    __restrict__ x2,
    unsigned long long* __restrict__ idxbuf,  // 1024, pre-set to ~0
    int N)
{
    __shared__ _Float16 Bs[2][32 * BROW];

    const int lane = threadIdx.x & 31;
    const int wave = threadIdx.x >> 5;
    const int rb0  = blockIdx.x * ROWS_PER_BLOCK + wave * 16;
    const bool active = rb0 < N;
    const int rowbase = active ? rb0 : 0;
    const int h   = lane >> 4;
    const int l16 = lane & 15;
    const int srow = threadIdx.x >> 3;
    const int sseg = (threadIdx.x & 7) * 32;

    v16h afrag[8];
    {
        const _Float16* arow = Ah + (size_t)(rowbase + l16) * DIM;
        #pragma unroll
        for (int kk = 0; kk < 8; kk++) {
            const int k0 = kk * 32 + h * 8;
            v16h f;
            *((v8h*)&f)       = *(const v8h*)(arow + k0);
            *(((v8h*)&f) + 1) = *(const v8h*)(arow + k0 + 16);
            afrag[kk] = f;
        }
    }

    float x2l[8];
    #pragma unroll
    for (int v = 0; v < 8; v++) x2l[v] = x2[rowbase + h * 8 + v];

    {
        const _Float16* src = Bh + (size_t)srow * DIM + sseg;
        _Float16* dst = &Bs[0][srow * BROW + sseg];
        #pragma unroll
        for (int cch = 0; cch < 4; cch++)
            *(v8h*)(dst + cch * 8) = *(const v8h*)(src + cch * 8);
    }
    __syncthreads();

    for (int r = 0; r < NROUND; r++) {
        const int cur = r & 1;
        const int base = r * 32;

        v8h g[4];
        if (r + 1 < NROUND) {
            const _Float16* src = Bh + (size_t)((r + 1) * 32 + srow) * DIM + sseg;
            #pragma unroll
            for (int cch = 0; cch < 4; cch++)
                g[cch] = *(const v8h*)(src + cch * 8);
        }

        const _Float16* bt0 = &Bs[cur][(unsigned)l16 * BROW + h * 8];
        const _Float16* bt1 = bt0 + 16 * BROW;
        v8f acc0 = {}, acc1 = {};
        v16h b0 = lds_bfrag(bt0, 0);
        v16h b1 = lds_bfrag(bt1, 0);
        #pragma unroll
        for (int kk = 0; kk < 8; kk++) {
            v16h n0, n1;
            if (kk < 7) {
                n0 = lds_bfrag(bt0, kk + 1);
                n1 = lds_bfrag(bt1, kk + 1);
            }
            acc0 = __builtin_amdgcn_wmma_f32_16x16x32_f16(
                false, afrag[kk], false, b0, (short)0, acc0, false, false);
            acc1 = __builtin_amdgcn_wmma_f32_16x16x32_f16(
                false, afrag[kk], false, b1, (short)0, acc1, false, false);
            b0 = n0; b1 = n1;
        }

        // per-lane min over its 8 rows for each subtile (rows increase with v)
        float bs0 = 3.4e38f, bs1 = 3.4e38f;
        int   bi0 = rowbase, bi1 = rowbase;
        #pragma unroll
        for (int v = 0; v < 8; v++) {
            float s0 = __builtin_fmaf(-2.0f, acc0[v], x2l[v]);
            float s1 = __builtin_fmaf(-2.0f, acc1[v], x2l[v]);
            if (s0 < bs0) { bs0 = s0; bi0 = rowbase + h * 8 + v; }
            if (s1 < bs1) { bs1 = s1; bi1 = rowbase + h * 8 + v; }
        }
        // combine halves: lane l <-> l^16 hold the same column, rows 0-7 / 8-15
        float os0 = __shfl_xor(bs0, 16, 32);
        int   oi0 = __shfl_xor(bi0, 16, 32);
        if (os0 < bs0 || (os0 == bs0 && oi0 < bi0)) { bs0 = os0; bi0 = oi0; }
        float os1 = __shfl_xor(bs1, 16, 32);
        int   oi1 = __shfl_xor(bi1, 16, 32);
        if (os1 < bs1 || (os1 == bs1 && oi1 < bi1)) { bs1 = os1; bi1 = oi1; }

        if (active && h == 0) {
            unsigned u0 = __float_as_uint(bs0);
            u0 = (u0 & 0x80000000u) ? ~u0 : (u0 | 0x80000000u);
            atomicMin(&idxbuf[base + l16],
                      ((unsigned long long)u0 << 32) | (unsigned)bi0);
            unsigned u1 = __float_as_uint(bs1);
            u1 = (u1 & 0x80000000u) ? ~u1 : (u1 | 0x80000000u);
            atomicMin(&idxbuf[base + 16 + l16],
                      ((unsigned long long)u1 << 32) | (unsigned)bi1);
        }

        if (r + 1 < NROUND) {
            _Float16* dst = &Bs[(r + 1) & 1][srow * BROW + sseg];
            #pragma unroll
            for (int cch = 0; cch < 4; cch++)
                *(v8h*)(dst + cch * 8) = g[cch];
        }
        __syncthreads();
    }
}

// ---------------------------------------------------------------------------
// Output writers
// ---------------------------------------------------------------------------
__global__ __launch_bounds__(256) void classes_out_kernel(
    const int* __restrict__ a, float* __restrict__ out, int N)
{
    const int i = blockIdx.x * 256 + threadIdx.x;
    if (i < N) out[i] = (float)a[i];
}

__global__ __launch_bounds__(256) void gather_kernel(
    const float* __restrict__ data, const unsigned long long* __restrict__ idxbuf,
    float* __restrict__ centers_out, float* __restrict__ index_out)
{
    const int j = blockIdx.x;
    const int k = threadIdx.x;
    const unsigned idx = (unsigned)(idxbuf[j] & 0xffffffffull);
    centers_out[(size_t)j * DIM + k] = data[(size_t)idx * DIM + k];
    if (k == 0) index_out[j] = (float)idx;
}

// ---------------------------------------------------------------------------
// Host orchestration (graph-capture safe: kernels + memsetAsync + memcpyAsync)
// ---------------------------------------------------------------------------
extern "C" void kernel_launch(void* const* d_in, const int* in_sizes, int n_in,
                              void* d_out, int out_size, void* d_ws, size_t ws_size,
                              hipStream_t stream) {
    const float* data = (const float*)d_in[0];
    const int N = in_sizes[0] / DIM;                 // 100000
    const int nblk = (N + ROWS_PER_BLOCK - 1) / ROWS_PER_BLOCK;

    char* p = (char*)d_ws;
    auto carve = [&](size_t bytes) {
        char* r = p;
        p += (bytes + 255) & ~(size_t)255;
        return (void*)r;
    };
    _Float16* dh      = (_Float16*)carve((size_t)N * DIM * sizeof(_Float16));   // 51.2 MB
    float*    x2      = (float*)   carve((size_t)N * sizeof(float));
    float*    c       = (float*)   carve((size_t)NCLUST * DIM * sizeof(float));
    _Float16* ch      = (_Float16*)carve((size_t)NCLUST * DIM * sizeof(_Float16));
    float*    c2      = (float*)   carve((size_t)NCLUST * sizeof(float));
    float*    sums    = (float*)   carve((size_t)NCLUST * DIM * sizeof(float));
    float*    counts  = (float*)   carve((size_t)NCLUST * sizeof(float));
    int*      assign  = (int*)     carve((size_t)N * sizeof(int));
    unsigned long long* idxbuf = (unsigned long long*)carve((size_t)NCLUST * 8);

    float* out = (float*)d_out;
    float* out_c       = out;                                   // 1024*256
    float* out_classes = out + (size_t)NCLUST * DIM;            // N
    float* out_centers = out_classes + N;                       // 1024*256
    float* out_index   = out_centers + (size_t)NCLUST * DIM;    // 1024

    preprocess_kernel<<<N, 256, 0, stream>>>(data, dh, x2);
    init_centers_kernel<<<NCLUST, 256, 0, stream>>>(data, c);

    for (int it = 0; it < NITER; it++) {
        prep_centers_kernel<<<NCLUST, 256, 0, stream>>>(c, ch, c2);
        hipMemsetAsync(sums, 0, (size_t)NCLUST * DIM * sizeof(float), stream);
        hipMemsetAsync(counts, 0, (size_t)NCLUST * sizeof(float), stream);
        assign_kernel<<<nblk, 256, 0, stream>>>(dh, ch, c2, assign, N);
        accumulate_kernel<<<N, 256, 0, stream>>>(data, assign, sums, counts);
        update_kernel<<<NCLUST, 256, 0, stream>>>(sums, counts, c);
    }

    hipMemcpyAsync(out_c, c, (size_t)NCLUST * DIM * sizeof(float),
                   hipMemcpyDeviceToDevice, stream);
    classes_out_kernel<<<(N + 255) / 256, 256, 0, stream>>>(assign, out_classes, N);

    prep_centers_kernel<<<NCLUST, 256, 0, stream>>>(c, ch, c2);
    hipMemsetAsync(idxbuf, 0xFF, (size_t)NCLUST * 8, stream);
    index_kernel<<<nblk, 256, 0, stream>>>(dh, ch, x2, idxbuf, N);
    gather_kernel<<<NCLUST, 256, 0, stream>>>(data, idxbuf, out_centers, out_index);
}
